// sarModule_13692355740377
// MI455X (gfx1250) — matrix operand order repaired and checked by reading, not compiled, not run on hardware
//
#include <hip/hip_runtime.h>

// ---------------------------------------------------------------------------
// CDNA5 (gfx1250) implementation of the SAR module.
// Heavy convolutions: implicit GEMM on v_wmma_f32_16x16x32_bf16 with the
// activation (B) tiles staged per-block into LDS through the CDNA5 async
// global->LDS path (global_load_async_to_lds_b128 / ASYNCcnt), double-buffered
// across the 3x3 taps so the async DMA overlaps WMMA compute.
// ---------------------------------------------------------------------------

typedef __bf16 bf16;
typedef __attribute__((ext_vector_type(16))) __bf16 bf16x16;
typedef __attribute__((ext_vector_type(8)))  __bf16 bf16x8;
typedef __attribute__((ext_vector_type(8)))  float   v8f;

static constexpr int BB = 16, CIN = 64, CO = 128;
static constexpr int HH = 112, WW = 112, HO = 56, WO = 56;
static constexpr int HP1 = HH + 2;   // 114 (padded 112)
static constexpr int HP0 = HO + 2;   // 58  (padded 56)

// ------------------------- workspace layout (bytes) ------------------------
static constexpr size_t XP_SZ   = (size_t)BB*HP1*HP1*CIN*2;  // x, padded NHWC bf16
static constexpr size_t AVG_SZ  = (size_t)BB*HP0*HP0*CIN*2;  // avgpool3s2(x)
static constexpr size_t H1_SZ   = (size_t)BB*HP0*HP0*CO*2;   // relu(bn(conv bb1))
static constexpr size_t RESB_SZ = (size_t)BB*HO*WO*CO*2;     // bn(conv1x1(avg))
static constexpr size_t XB_SZ   = H1_SZ;                     // x_base
static constexpr size_t G_SZ    = (size_t)BB*4*HO*WO*4;      // maskgen feature (f32)
static constexpr size_t MSK_SZ  = 4096;                      // mask 16x7x7 f32
static constexpr size_t XG_SZ   = XP_SZ;                     // gated x
static constexpr size_t R1_SZ   = (size_t)BB*HP1*HP1*CO*2;   // refine mid
static constexpr size_t R2_SZ   = (size_t)BB*HH*WW*CO*2;     // refine conv2 (unpadded)
static constexpr size_t RR_SZ   = R2_SZ;                     // refine shortcut

static constexpr size_t XP_OFF   = 0;
static constexpr size_t AVG_OFF  = XP_OFF   + XP_SZ;
static constexpr size_t H1_OFF   = AVG_OFF  + AVG_SZ;
static constexpr size_t RESB_OFF = H1_OFF   + H1_SZ;
static constexpr size_t XB_OFF   = RESB_OFF + RESB_SZ;
static constexpr size_t G_OFF    = XB_OFF   + XB_SZ;
static constexpr size_t MSK_OFF  = G_OFF    + G_SZ;
static constexpr size_t XG_OFF   = MSK_OFF  + MSK_SZ;
static constexpr size_t R1_OFF   = XG_OFF   + XG_SZ;
static constexpr size_t R2_OFF   = R1_OFF   + R1_SZ;
static constexpr size_t RR_OFF   = R2_OFF   + R2_SZ;
static constexpr size_t PA_OFF   = RR_OFF   + RR_SZ;
// lifetime-disjoint aliases:
static constexpr size_t OUTP_OFF = XG_OFF;   // overlays XG+R1 (both dead by combine)
static constexpr size_t OUTP_SZ  = R1_SZ;    // padded 114x114x128 bf16
static constexpr size_t F1_OFF   = XP_OFF;   // overlays XP (dead after rb_dw conv)

// packed weights (A fragments): bytes per conv = 8 tiles * chunks * 32 lanes * 16 * 2B
static constexpr size_t PA_BB1 = PA_OFF;                   // 3x3, Cin64 : 18 chunks
static constexpr size_t PA_BDW = PA_BB1 + (size_t)18*8192; // 1x1, Cin64 : 2
static constexpr size_t PA_BB2 = PA_BDW + (size_t)2*8192;  // 3x3, Cin128: 36
static constexpr size_t PA_RB1 = PA_BB2 + (size_t)36*8192; // 3x3, Cin64 : 18
static constexpr size_t PA_RDW = PA_RB1 + (size_t)18*8192; // 1x1, Cin64 : 2
static constexpr size_t PA_RB2 = PA_RDW + (size_t)2*8192;  // 3x3, Cin128: 36
static constexpr size_t PA_FU1 = PA_RB2 + (size_t)36*8192; // 3x3, Cin128: 36
static constexpr size_t PA_FU2 = PA_FU1 + (size_t)36*8192; // 3x3, Cin128: 36

static __device__ __forceinline__ v8f wmma_bf16(bf16x16 a, bf16x16 b, v8f c) {
  return __builtin_amdgcn_wmma_f32_16x16x32_bf16(false, a, false, b, (short)0, c,
                                                 false, false);
}

// ---------------------------------------------------------------------------
// Weight packing: fold BN scale, swizzle into WMMA A-fragment (16x32 bf16)
// lane order.  K ordering: k = (r*ks+s)*Cin + ci  (tap-major, channel-minor).
// Per A-table: lane l (g=l/16, m=l%16), element j in [0,16):
//   k_local = (j&7) + 8*g + 16*(j>>3)
// ---------------------------------------------------------------------------
__global__ void k_pack(const float* __restrict__ w, const float* __restrict__ sc,
                       bf16* __restrict__ pa, int Cin, int ks, int chunks) {
  int idx = blockIdx.x * blockDim.x + threadIdx.x;
  int total = 8 * chunks * 512;
  if (idx >= total) return;
  int j    = idx & 15;
  int lane = (idx >> 4) & 31;
  int kk   = (idx >> 9) % chunks;
  int tile = idx / (chunks * 512);
  int m = lane & 15, g = lane >> 4;
  int co = tile * 16 + m;
  int kl = (j & 7) + 8 * g + 16 * (j >> 3);
  int k  = kk * 32 + kl;
  int tap = k / Cin, ci = k % Cin;
  int r = tap / ks, s = tap % ks;
  float v = w[(((size_t)co * Cin + ci) * ks + r) * ks + s] * sc[co];
  pa[idx] = (bf16)v;
}

// NCHW f32 -> padded NHWC bf16 (halo pre-zeroed by memset)
__global__ void k_transform(const float* __restrict__ x, bf16* __restrict__ dst) {
  size_t idx = (size_t)blockIdx.x * blockDim.x + threadIdx.x;
  if (idx >= (size_t)BB * HH * WW * CIN) return;
  int ci = idx & 63;
  size_t p = idx >> 6;
  int w = (int)(p % WW); p /= WW;
  int h = (int)(p % HH);
  int b = (int)(p / HH);
  float v = x[(((size_t)b * CIN + ci) * HH + h) * WW + w];
  dst[(((size_t)b * HP1 + h + 1) * HP1 + (w + 1)) * CIN + ci] = (bf16)v;
}

// gated transform: XG = x * mask_upsampled (nearest 16x)
__global__ void k_gate(const float* __restrict__ x, const float* __restrict__ msk,
                       bf16* __restrict__ dst) {
  size_t idx = (size_t)blockIdx.x * blockDim.x + threadIdx.x;
  if (idx >= (size_t)BB * HH * WW * CIN) return;
  int ci = idx & 63;
  size_t p = idx >> 6;
  int w = (int)(p % WW); p /= WW;
  int h = (int)(p % HH);
  int b = (int)(p / HH);
  float m = msk[b * 49 + (h >> 4) * 7 + (w >> 4)];
  float v = x[(((size_t)b * CIN + ci) * HH + h) * WW + w] * m;
  dst[(((size_t)b * HP1 + h + 1) * HP1 + (w + 1)) * CIN + ci] = (bf16)v;
}

// avgpool 3x3 stride2 pad1 (count_include_pad): padded NHWC -> padded NHWC, 64ch
__global__ void k_avgpool64(const bf16* __restrict__ src, bf16* __restrict__ dst) {
  size_t idx = (size_t)blockIdx.x * blockDim.x + threadIdx.x;
  if (idx >= (size_t)BB * HO * WO * CIN) return;
  int ci = idx & 63;
  size_t p = idx >> 6;
  int w = (int)(p % WO); p /= WO;
  int h = (int)(p % HO);
  int b = (int)(p / HO);
  float s = 0.f;
  for (int dy = 0; dy < 3; ++dy)
    for (int dx = 0; dx < 3; ++dx)
      s += (float)src[(((size_t)b * HP1 + 2 * h + dy) * HP1 + 2 * w + dx) * CIN + ci];
  dst[(((size_t)b * HP0 + h + 1) * HP0 + (w + 1)) * CIN + ci] = (bf16)(s * (1.f / 9.f));
}

// ---------------------------------------------------------------------------
// Implicit-GEMM convolution on WMMA with async-LDS staging.
// Block = 256 threads = 8 waves; block tile = 128 channels x 64 flat pixels.
// Per tap: the 64-pixel x Cin activation slab is DMA'd into LDS once
// (global_load_async_to_lds_b128, ASYNCcnt) and shared by the 4 channel-waves;
// double-buffered across taps so the DMA overlaps WMMA compute.
// MODE: 0 plain, 1 relu, 2 residual(bf16 NHWC)+relu,
//       3 final (avgpool3s2 shortcut + relu + f32 NCHW).
// CINCH = Cin/32 (2 or 4).
// ---------------------------------------------------------------------------
struct ConvArgs {
  const bf16* in; const bf16* pa; const float* bias;
  bf16* outb; const bf16* resid; float* outf; const bf16* avgsrc;
  int Hp, Wp, stride, ks, padAdj, chunks;
  int oHp, oWp, oOff, Hout, Wout;
};

template <int MODE, int CINCH>
__global__ __launch_bounds__(256) void k_conv(ConvArgs A) {
  constexpr int CINel = CINCH * 32;        // input channels
  constexpr int UPP   = CINCH * 4;         // 16B units per pixel row
  constexpr int ROWB  = CINCH * 64 + 16;   // LDS bytes per pixel row (+16 pad)
  constexpr int BUFB  = 64 * ROWB;         // one tap buffer
  __shared__ __align__(32) char smem[2][BUFB];

  const int tid  = threadIdx.x;
  const int lane = tid & 31;
  const int wv   = tid >> 5;
  const int cw   = wv & 3;        // channel tile-pair 0..3 -> co base 32*cw
  const int pw   = wv >> 2;       // pixel half 0..1
  const int g    = lane >> 4;
  const int ln   = lane & 15;
  const int coB  = cw * 32;
  const int HWp  = A.Hout * A.Wout;

  // ---- compute-fragment pixel coordinates (per lane, 2 pixel tiles) -------
  int pb[2], ph[2], pwd[2], pixL[2];
  size_t P[2];
#pragma unroll
  for (int t = 0; t < 2; ++t) {
    pixL[t] = pw * 32 + t * 16 + ln;
    size_t p = (size_t)blockIdx.x * 64 + pixL[t];
    P[t] = p;
    int b = (int)(p / HWp);
    int r = (int)(p - (size_t)b * HWp);
    pb[t] = b; ph[t] = r / A.Wout; pwd[t] = r % A.Wout;
  }

  // ---- staging-unit coordinates (per thread, CINCH x 16B units) -----------
  int ub[CINCH], uh[CINCH], uw[CINCH], upart[CINCH];
  unsigned uoff[CINCH];
#pragma unroll
  for (int i = 0; i < CINCH; ++i) {
    int u   = tid + i * 256;
    int pix = u / UPP;
    int part = u - pix * UPP;
    size_t p = (size_t)blockIdx.x * 64 + pix;
    int b = (int)(p / HWp);
    int r = (int)(p - (size_t)b * HWp);
    ub[i] = b; uh[i] = r / A.Wout; uw[i] = r % A.Wout;
    upart[i] = part;
    uoff[i]  = (unsigned)(pix * ROWB + part * 16);
  }

  auto stage = [&](int tap, int sel) {
    const int r = tap / A.ks, s = tap - (tap / A.ks) * A.ks;
#pragma unroll
    for (int i = 0; i < CINCH; ++i) {
      const bf16* gp = A.in +
          (((size_t)ub[i] * A.Hp + uh[i] * A.stride + r + A.padAdj) * A.Wp +
           (uw[i] * A.stride + s + A.padAdj)) * CINel + upart[i] * 8;
      unsigned laddr = (unsigned)(size_t)(&smem[sel][0]) + uoff[i];
      // CDNA5 async DMA: 16B global -> LDS, tracked with ASYNCcnt
      asm volatile("global_load_async_to_lds_b128 %0, %1, off"
                   :: "v"(laddr), "v"(gp) : "memory");
    }
  };

  union BU { bf16x16 v; bf16x8 h[2]; };

  v8f acc[2][2] = {};
  const int taps = A.ks * A.ks;

  stage(0, 0);
  asm volatile("s_wait_asynccnt 0x0" ::: "memory");
  __syncthreads();

  for (int tap = 0; tap < taps; ++tap) {
    const int sel = tap & 1;
    if (tap + 1 < taps) stage(tap + 1, sel ^ 1);

    const bf16* ap0 = A.pa + ((size_t)((cw * 2 + 0) * A.chunks + tap * CINCH) * 32 + lane) * 16;
    const bf16* ap1 = A.pa + ((size_t)((cw * 2 + 1) * A.chunks + tap * CINCH) * 32 + lane) * 16;
    __builtin_prefetch(ap0 + (size_t)CINCH * 512, 0, 1);  // next tap's weights
    const char* bufp = &smem[sel][0];
    const char* bp0 = bufp + pixL[0] * ROWB + g * 16;
    const char* bp1 = bufp + pixL[1] * ROWB + g * 16;
#pragma unroll
    for (int kc = 0; kc < CINCH; ++kc) {
      bf16x16 a0 = *(const bf16x16*)(ap0 + (size_t)kc * 512);
      bf16x16 a1 = *(const bf16x16*)(ap1 + (size_t)kc * 512);
      BU b0, b1;
      b0.h[0] = *(const bf16x8*)(bp0 + kc * 64);
      b0.h[1] = *(const bf16x8*)(bp0 + kc * 64 + 32);
      b1.h[0] = *(const bf16x8*)(bp1 + kc * 64);
      b1.h[1] = *(const bf16x8*)(bp1 + kc * 64 + 32);
      acc[0][0] = wmma_bf16(a0, b0.v, acc[0][0]);
      acc[0][1] = wmma_bf16(a0, b1.v, acc[0][1]);
      acc[1][0] = wmma_bf16(a1, b0.v, acc[1][0]);
      acc[1][1] = wmma_bf16(a1, b1.v, acc[1][1]);
    }

    if (tap + 1 < taps) {
      asm volatile("s_wait_asynccnt 0x0" ::: "memory");
      __syncthreads();
    }
  }

  // ---------------- epilogue: bias (+residual/avgpool) (+relu), store -------
#pragma unroll
  for (int i = 0; i < 2; ++i) {
    const int co = coB + i * 16 + g * 8;   // 8 contiguous output channels
    float b8[8];
#pragma unroll
    for (int e = 0; e < 8; ++e) b8[e] = A.bias[co + e];
#pragma unroll
    for (int t = 0; t < 2; ++t) {
      float v[8];
#pragma unroll
      for (int e = 0; e < 8; ++e) v[e] = acc[i][t][e] + b8[e];
      if constexpr (MODE == 2) {
        bf16x8 rv = *(const bf16x8*)(A.resid + P[t] * CO + co);
#pragma unroll
        for (int e = 0; e < 8; ++e) v[e] += (float)rv[e];
      }
      if constexpr (MODE == 3) {
        float s9[8];
#pragma unroll
        for (int e = 0; e < 8; ++e) s9[e] = 0.f;
        for (int dy = 0; dy < 3; ++dy)
          for (int dx = 0; dx < 3; ++dx) {
            bf16x8 q = *(const bf16x8*)(A.avgsrc +
                (((size_t)pb[t] * HP1 + ph[t] * 2 + dy) * HP1 + pwd[t] * 2 + dx) * CO + co);
#pragma unroll
            for (int e = 0; e < 8; ++e) s9[e] += (float)q[e];
          }
#pragma unroll
        for (int e = 0; e < 8; ++e) v[e] += s9[e] * (1.f / 9.f);
      }
      if constexpr (MODE >= 1) {
#pragma unroll
        for (int e = 0; e < 8; ++e) v[e] = fmaxf(v[e], 0.f);
      }
      if constexpr (MODE == 3) {
#pragma unroll
        for (int e = 0; e < 8; ++e)
          A.outf[(((size_t)pb[t] * CO + co + e) * HO + ph[t]) * WO + pwd[t]] = v[e];
      } else {
        bf16x8 o;
#pragma unroll
        for (int e = 0; e < 8; ++e) o[e] = (bf16)v[e];
        *(bf16x8*)(A.outb +
            (((size_t)pb[t] * A.oHp + ph[t] + A.oOff) * A.oWp + pwd[t] + A.oOff) * CO + co) = o;
      }
    }
  }
}

// ---------------------------------------------------------------------------
// maskGen: 3x3 conv 128->4 + bn + relu (plain VALU, 0.46 GFLOP)
// ---------------------------------------------------------------------------
__global__ void k_mgconv(const bf16* __restrict__ xb, const float* __restrict__ mgw,
                         const float* __restrict__ mgs, const float* __restrict__ mgb,
                         float* __restrict__ G) {
  int idx = blockIdx.x * blockDim.x + threadIdx.x;
  if (idx >= BB * 4 * HO * WO) return;
  int w = idx % WO;
  int h = (idx / WO) % HO;
  int c = (idx / (HO * WO)) % 4;
  int b = idx / (4 * HO * WO);
  float acc = 0.f;
  for (int r = 0; r < 3; ++r)
    for (int s = 0; s < 3; ++s) {
      const bf16* p = xb + (((size_t)b * HP0 + h + r) * HP0 + (w + s)) * CO;
      const float* wp = mgw + ((size_t)c * CO) * 9 + r * 3 + s;
      for (int ci = 0; ci < CO; ++ci) acc += (float)p[ci] * wp[(size_t)ci * 9];
    }
  G[idx] = fmaxf(acc * mgs[c] + mgb[c], 0.f);
}

// adaptive avg-pool 7x7 + 1x1 fc + hard threshold -> mask (ws copy + d_out tail)
__global__ void k_mask(const float* __restrict__ G, const float* __restrict__ fcw,
                       const float* __restrict__ fcb, float* __restrict__ msk,
                       float* __restrict__ outmsk) {
  int idx = blockIdx.x * blockDim.x + threadIdx.x;
  if (idx >= BB * 49) return;
  int j = idx % 7, i = (idx / 7) % 7, b = idx / 49;
  float v = fcb[0];
  for (int c = 0; c < 4; ++c) {
    float s = 0.f;
    for (int y = 0; y < 8; ++y)
      for (int x = 0; x < 8; ++x)
        s += G[(((size_t)b * 4 + c) * HO + i * 8 + y) * WO + j * 8 + x];
    v += fcw[c] * (s * (1.f / 64.f));
  }
  float m = (v >= 0.f) ? 1.f : 0.f;
  msk[idx] = m;
  outmsk[idx] = m;
}

// out_pre = relu(bilinear2x(x_base) + R2*mask + RRES), written padded NHWC bf16
__global__ void k_combine(const bf16* __restrict__ XB, const bf16* __restrict__ R2,
                          const bf16* __restrict__ RR, const float* __restrict__ msk,
                          bf16* __restrict__ OUTP) {
  size_t idx = (size_t)blockIdx.x * blockDim.x + threadIdx.x;
  if (idx >= (size_t)BB * HH * WW * 16) return;
  int cg = (int)(idx & 15);
  size_t p = idx >> 4;
  int w = (int)(p % WW); p /= WW;
  int h = (int)(p % HH);
  int b = (int)(p / HH);
  int co = cg * 8;
  float m = msk[b * 49 + (h >> 4) * 7 + (w >> 4)];
  bf16x8 r2 = *(const bf16x8*)(R2 + (((size_t)b * HH + h) * WW + w) * CO + co);
  bf16x8 rr = *(const bf16x8*)(RR + (((size_t)b * HH + h) * WW + w) * CO + co);
  // bilinear, align_corners=False, scale 2: src = 0.5*i - 0.25
  int h0, h1, w0, w1; float wh1, ww1;
  if (h & 1) { h0 = h >> 1; h1 = h0 + 1; wh1 = 0.25f; }
  else       { h1 = h >> 1; h0 = h1 - 1; wh1 = 0.75f; }
  if (w & 1) { w0 = w >> 1; w1 = w0 + 1; ww1 = 0.25f; }
  else       { w1 = w >> 1; w0 = w1 - 1; ww1 = 0.75f; }
  h0 = h0 < 0 ? 0 : h0;  h1 = h1 > HO - 1 ? HO - 1 : h1;
  w0 = w0 < 0 ? 0 : w0;  w1 = w1 > WO - 1 ? WO - 1 : w1;
  float wh0 = 1.f - wh1, ww0 = 1.f - ww1;
  bf16x8 q00 = *(const bf16x8*)(XB + (((size_t)b * HP0 + h0 + 1) * HP0 + w0 + 1) * CO + co);
  bf16x8 q01 = *(const bf16x8*)(XB + (((size_t)b * HP0 + h0 + 1) * HP0 + w1 + 1) * CO + co);
  bf16x8 q10 = *(const bf16x8*)(XB + (((size_t)b * HP0 + h1 + 1) * HP0 + w0 + 1) * CO + co);
  bf16x8 q11 = *(const bf16x8*)(XB + (((size_t)b * HP0 + h1 + 1) * HP0 + w1 + 1) * CO + co);
  bf16x8 o;
#pragma unroll
  for (int e = 0; e < 8; ++e) {
    float bil = wh0 * (ww0 * (float)q00[e] + ww1 * (float)q01[e]) +
                wh1 * (ww0 * (float)q10[e] + ww1 * (float)q11[e]);
    float v = bil + (float)r2[e] * m + (float)rr[e];
    o[e] = (bf16)fmaxf(v, 0.f);
  }
  *(bf16x8*)(OUTP + (((size_t)b * HP1 + h + 1) * HP1 + (w + 1)) * CO + co) = o;
}

// ---------------------------------------------------------------------------
extern "C" void kernel_launch(void* const* d_in, const int* in_sizes, int n_in,
                              void* d_out, int out_size, void* d_ws, size_t ws_size,
                              hipStream_t stream) {
  (void)in_sizes; (void)n_in; (void)out_size; (void)ws_size;
  const float* x      = (const float*)d_in[0];
  const float* bb_w1  = (const float*)d_in[1];
  const float* bb_s1  = (const float*)d_in[2];
  const float* bb_b1  = (const float*)d_in[3];
  const float* bb_w2  = (const float*)d_in[4];
  const float* bb_s2  = (const float*)d_in[5];
  const float* bb_b2  = (const float*)d_in[6];
  const float* bb_dw  = (const float*)d_in[7];
  const float* bb_ds  = (const float*)d_in[8];
  const float* bb_db  = (const float*)d_in[9];
  const float* rb_w1  = (const float*)d_in[10];
  const float* rb_s1  = (const float*)d_in[11];
  const float* rb_b1  = (const float*)d_in[12];
  const float* rb_w2  = (const float*)d_in[13];
  const float* rb_s2  = (const float*)d_in[14];
  const float* rb_b2  = (const float*)d_in[15];
  const float* rb_dw  = (const float*)d_in[16];
  const float* rb_ds  = (const float*)d_in[17];
  const float* rb_db  = (const float*)d_in[18];
  const float* mg_w   = (const float*)d_in[19];
  const float* mg_s   = (const float*)d_in[20];
  const float* mg_b   = (const float*)d_in[21];
  const float* mg_fcw = (const float*)d_in[22];
  const float* mg_fcb = (const float*)d_in[23];
  const float* fu_w1  = (const float*)d_in[24];
  const float* fu_s1  = (const float*)d_in[25];
  const float* fu_b1  = (const float*)d_in[26];
  const float* fu_w2  = (const float*)d_in[27];
  const float* fu_s2  = (const float*)d_in[28];
  const float* fu_b2  = (const float*)d_in[29];

  float* out = (float*)d_out;
  float* outmask = out + (size_t)BB * CO * HO * WO;
  char* ws = (char*)d_ws;

  bf16*  XP   = (bf16*)(ws + XP_OFF);
  bf16*  AVG  = (bf16*)(ws + AVG_OFF);
  bf16*  H1   = (bf16*)(ws + H1_OFF);
  bf16*  RESB = (bf16*)(ws + RESB_OFF);
  bf16*  XB   = (bf16*)(ws + XB_OFF);
  float* G    = (float*)(ws + G_OFF);
  float* MSK  = (float*)(ws + MSK_OFF);
  bf16*  XG   = (bf16*)(ws + XG_OFF);
  bf16*  R1   = (bf16*)(ws + R1_OFF);
  bf16*  R2   = (bf16*)(ws + R2_OFF);
  bf16*  RR   = (bf16*)(ws + RR_OFF);
  bf16*  OUTP = (bf16*)(ws + OUTP_OFF);
  bf16*  F1   = (bf16*)(ws + F1_OFF);

  const int TPB = 256;
  const size_t nelemX = (size_t)BB * HH * WW * CIN;      // 12.8M
  const int gridX  = (int)((nelemX + TPB - 1) / TPB);
  const int gridP0 = BB * HO * WO / 64;                  // 784  (56x56 convs)
  const int gridP1 = BB * HH * WW / 64;                  // 3136 (112x112 convs)

  // ---- stage 0: transforms + weight packing
  hipMemsetAsync(ws + XP_OFF, 0, XP_SZ, stream);
  k_transform<<<gridX, TPB, 0, stream>>>(x, XP);
  k_pack<<<(8*18*512 + TPB-1)/TPB, TPB, 0, stream>>>(bb_w1, bb_s1, (bf16*)(ws+PA_BB1), CIN, 3, 18);
  k_pack<<<(8* 2*512 + TPB-1)/TPB, TPB, 0, stream>>>(bb_dw, bb_ds, (bf16*)(ws+PA_BDW), CIN, 1, 2);
  k_pack<<<(8*36*512 + TPB-1)/TPB, TPB, 0, stream>>>(bb_w2, bb_s2, (bf16*)(ws+PA_BB2), CO, 3, 36);
  k_pack<<<(8*18*512 + TPB-1)/TPB, TPB, 0, stream>>>(rb_w1, rb_s1, (bf16*)(ws+PA_RB1), CIN, 3, 18);
  k_pack<<<(8* 2*512 + TPB-1)/TPB, TPB, 0, stream>>>(rb_dw, rb_ds, (bf16*)(ws+PA_RDW), CIN, 1, 2);
  k_pack<<<(8*36*512 + TPB-1)/TPB, TPB, 0, stream>>>(rb_w2, rb_s2, (bf16*)(ws+PA_RB2), CO, 3, 36);
  k_pack<<<(8*36*512 + TPB-1)/TPB, TPB, 0, stream>>>(fu_w1, fu_s1, (bf16*)(ws+PA_FU1), CO, 3, 36);
  k_pack<<<(8*36*512 + TPB-1)/TPB, TPB, 0, stream>>>(fu_w2, fu_s2, (bf16*)(ws+PA_FU2), CO, 3, 36);
  k_avgpool64<<<(int)(((size_t)BB*HO*WO*CIN + TPB-1)/TPB), TPB, 0, stream>>>(XP, AVG);

  // ---- base branch
  hipMemsetAsync(ws + H1_OFF, 0, H1_SZ, stream);
  { ConvArgs a{XP, (bf16*)(ws+PA_BB1), bb_b1, H1, nullptr, nullptr, nullptr,
               HP1, HP1, 2, 3, 0, 18, HP0, HP0, 1, HO, WO};
    k_conv<1, 2><<<gridP0, TPB, 0, stream>>>(a); }              // bb1: relu
  { ConvArgs a{AVG, (bf16*)(ws+PA_BDW), bb_db, RESB, nullptr, nullptr, nullptr,
               HP0, HP0, 1, 1, 1, 2, HO, WO, 0, HO, WO};
    k_conv<0, 2><<<gridP0, TPB, 0, stream>>>(a); }              // 1x1 shortcut
  hipMemsetAsync(ws + XB_OFF, 0, XB_SZ, stream);
  { ConvArgs a{H1, (bf16*)(ws+PA_BB2), bb_b2, XB, RESB, nullptr, nullptr,
               HP0, HP0, 1, 3, 0, 36, HP0, HP0, 1, HO, WO};
    k_conv<2, 4><<<gridP0, TPB, 0, stream>>>(a); }              // bb2: +res, relu

  // ---- mask generation
  k_mgconv<<<(BB*4*HO*WO + TPB-1)/TPB, TPB, 0, stream>>>(XB, mg_w, mg_s, mg_b, G);
  k_mask<<<(BB*49 + TPB-1)/TPB, TPB, 0, stream>>>(G, mg_fcw, mg_fcb, MSK, outmask);

  // ---- refine branch
  hipMemsetAsync(ws + XG_OFF, 0, XG_SZ, stream);
  k_gate<<<gridX, TPB, 0, stream>>>(x, MSK, XG);
  hipMemsetAsync(ws + R1_OFF, 0, R1_SZ, stream);
  { ConvArgs a{XG, (bf16*)(ws+PA_RB1), rb_b1, R1, nullptr, nullptr, nullptr,
               HP1, HP1, 1, 3, 0, 18, HP1, HP1, 1, HH, WW};
    k_conv<1, 2><<<gridP1, TPB, 0, stream>>>(a); }              // rb1: relu
  { ConvArgs a{XP, (bf16*)(ws+PA_RDW), rb_db, RR, nullptr, nullptr, nullptr,
               HP1, HP1, 1, 1, 1, 2, HH, WW, 0, HH, WW};
    k_conv<0, 2><<<gridP1, TPB, 0, stream>>>(a); }              // 1x1 shortcut
  { ConvArgs a{R1, (bf16*)(ws+PA_RB2), rb_b2, R2, nullptr, nullptr, nullptr,
               HP1, HP1, 1, 3, 0, 36, HH, WW, 0, HH, WW};
    k_conv<0, 4><<<gridP1, TPB, 0, stream>>>(a); }              // rb2: raw

  // ---- combine: relu(upsample(x_base) + R2*mask + RRES)  (OUTP overlays XG+R1)
  hipMemsetAsync(ws + OUTP_OFF, 0, OUTP_SZ, stream);
  k_combine<<<(int)(((size_t)BB*HH*WW*16 + TPB-1)/TPB), TPB, 0, stream>>>(XB, R2, RR, MSK, OUTP);

  // ---- fusion block (F1 overlays XP, now dead)
  hipMemsetAsync(ws + F1_OFF, 0, H1_SZ, stream);
  { ConvArgs a{OUTP, (bf16*)(ws+PA_FU1), fu_b1, F1, nullptr, nullptr, nullptr,
               HP1, HP1, 2, 3, 0, 36, HP0, HP0, 1, HO, WO};
    k_conv<1, 4><<<gridP0, TPB, 0, stream>>>(a); }              // fu1: relu
  { ConvArgs a{F1, (bf16*)(ws+PA_FU2), fu_b2, nullptr, nullptr, out, OUTP,
               HP0, HP0, 1, 3, 0, 36, 0, 0, 0, HO, WO};
    k_conv<3, 4><<<gridP0, TPB, 0, stream>>>(a); }              // fu2: +avgpool, relu, f32 NCHW
}
